// GATMultiLayer_64561948393754
// MI455X (gfx1250) — matrix operand order repaired and checked by reading, not compiled, not run on hardware
//
#include <hip/hip_runtime.h>
#include <hip/hip_bf16.h>
#include <math.h>

typedef __attribute__((ext_vector_type(16))) __bf16   v16bf;
typedef __attribute__((ext_vector_type(8)))  float    v8f;
typedef __attribute__((ext_vector_type(4)))  unsigned u32x4;

#define DEVFN __device__ __forceinline__

namespace {

constexpr int N_NODES = 65536;          // B*L
constexpr int E_EDGES = 65535;          // N-1
constexpr int TE      = E_EDGES + N_NODES; // edges + self loops = 131071
constexpr int D_IN    = 512;
constexpr int D_OUT   = 256;
constexpr int HEADS   = 4;
constexpr int K_DIM   = 2 * D_IN + D_OUT;   // 1280
constexpr int N_OUT   = HEADS * D_OUT;      // 1024
constexpr int NKB     = K_DIM / 32;         // 40 K-steps
constexpr float SLOPE = 0.2f;

struct SelState {
  unsigned nnz;
  unsigned doPrune;
  unsigned kRem;
  unsigned prefix;
  unsigned hist[256];
};

// ---- workspace layout (bytes) ----
constexpr size_t OFF_H      = 0;                                            // N*1024 f32
constexpr size_t OFF_WB     = OFF_H      + (size_t)N_NODES * N_OUT * 4;     // N_OUT*K_DIM bf16 (W^T)
constexpr size_t OFF_SCORE  = OFF_WB     + (size_t)K_DIM * N_OUT * 2;       // N f32
constexpr size_t OFF_ASRC   = OFF_SCORE  + (size_t)N_NODES * 4;             // N*4 f32
constexpr size_t OFF_ADST   = OFF_ASRC   + (size_t)N_NODES * HEADS * 4;     // N*4 f32
constexpr size_t OFF_M      = OFF_ADST   + (size_t)N_NODES * HEADS * 4;     // N*4 u32 (encoded max)
constexpr size_t OFF_S      = OFF_M      + (size_t)N_NODES * HEADS * 4;     // N*4 f32
constexpr size_t OFF_ALPHA  = OFF_S      + (size_t)N_NODES * HEADS * 4;     // TE*4 f32
constexpr size_t OFF_REMOVE = OFF_ALPHA  + ((size_t)TE * HEADS * 4 + 16);   // N int
constexpr size_t OFF_P0     = OFF_REMOVE + (size_t)N_NODES * 4;             // E int
constexpr size_t OFF_P1     = OFF_P0     + (size_t)(E_EDGES + 1) * 4;       // E int
constexpr size_t OFF_EMB    = OFF_P1     + (size_t)(E_EDGES + 1) * 4;       // 2*256 f32
constexpr size_t OFF_STATE  = OFF_EMB    + 512 * 4;

DEVFN float geluf(float v) {
  return 0.5f * v * (1.0f + tanhf(0.7978845608028654f * (v + 0.044715f * v * v * v)));
}
DEVFN unsigned f2bf(float f) {
  unsigned u = __float_as_uint(f);
  unsigned r = u + 0x7FFFu + ((u >> 16) & 1u);   // round-to-nearest-even
  return r >> 16;
}
DEVFN unsigned encf(float f) {                    // order-preserving float->uint
  unsigned u = __float_as_uint(f);
  return (u >> 31) ? ~u : (u | 0x80000000u);
}
DEVFN float decf(unsigned u) {
  unsigned b = (u >> 31) ? (u ^ 0x80000000u) : ~u;
  return __uint_as_float(b);
}
constexpr unsigned ENC_NEG_INF = 0x007FFFFFu;     // encf(-inf)

// ---------------- init kernels ----------------
__global__ void k_zero_out(float* out, long long n) {
  long long i = (long long)blockIdx.x * blockDim.x + threadIdx.x;
  long long st = (long long)gridDim.x * blockDim.x;
  for (; i < n; i += st) out[i] = 0.0f;
}
__global__ void k_init_ms(unsigned* m, float* s) {
  int i = blockIdx.x * blockDim.x + threadIdx.x;
  if (i < N_NODES * HEADS) { m[i] = ENC_NEG_INF; s[i] = 0.0f; }
}
__global__ void k_init_state(SelState* st) {
  int t = threadIdx.x;
  if (t == 0) { st->nnz = 0; st->doPrune = 0; st->kRem = 0; st->prefix = 0; }
  st->hist[t] = 0;
}
__global__ void k_emb_tab(const float* emb_w, const float* emb_b, float* tab) {
  int t = threadIdx.x;                 // 512 threads
  int c = t & (D_OUT - 1);
  tab[t] = geluf(emb_w[t] + emb_b[c]);
}
// transpose weights to [n][k] bf16 so B tiles are row-contiguous for async DMA
__global__ void k_prep_w(const float* gw, unsigned short* wbT) {
  int i = blockIdx.x * blockDim.x + threadIdx.x;
  if (i < K_DIM * N_OUT) {
    int k = i / N_OUT, n = i - k * N_OUT;
    wbT[(size_t)n * K_DIM + k] = (unsigned short)f2bf(gw[i]);
  }
}

// ---------------- node scoring ----------------
__global__ __launch_bounds__(256) void k_score(const float* __restrict__ x,
                                               const float* __restrict__ sr,
                                               const float* __restrict__ scw,
                                               const float* __restrict__ scb,
                                               const float* __restrict__ nm,
                                               float* score, SelState* st) {
  __shared__ float red[256];
  int n = blockIdx.x, t = threadIdx.x;
  float p = 0.0f;
  for (int i = 0; i < 2; ++i) {
    int j = t + 256 * i;
    p += x[(size_t)n * D_IN + j] * scw[j];
    p += sr[(size_t)n * D_IN + j] * scw[D_IN + j];
  }
  red[t] = p; __syncthreads();
  for (int off = 128; off > 0; off >>= 1) {
    if (t < off) red[t] += red[t + off];
    __syncthreads();
  }
  if (t == 0) {
    float sc = (red[0] + scb[0]) * nm[n];
    score[n] = sc;
    if (sc != 0.0f) atomicAdd(&st->nnz, 1u);
  }
}

__global__ void k_compute_k(SelState* st) {
  double kf = (double)st->nnz * 0.3;
  if ((int)kf != 0) { st->doPrune = 1; st->kRem = (unsigned)(kf + 1.0); }
  else st->doPrune = 0;
  st->prefix = 0;
}

__global__ void k_hist(const float* score, SelState* st, int round) {
  int i = blockIdx.x * blockDim.x + threadIdx.x;
  if (i >= N_NODES || !st->doPrune) return;
  float sc = score[i];
  if (sc == 0.0f) return;
  unsigned u = encf(sc);
  int shift = 24 - 8 * round;
  if (round > 0 && (u >> (shift + 8)) != st->prefix) return;
  atomicAdd(&st->hist[(u >> shift) & 0xFFu], 1u);
}

__global__ void k_scan(SelState* st) {
  __shared__ unsigned c[256];
  int t = threadIdx.x;
  c[t] = st->hist[t]; __syncthreads();
  for (int off = 1; off < 256; off <<= 1) {
    unsigned v = (t >= off) ? c[t - off] : 0u;
    __syncthreads();
    c[t] += v;
    __syncthreads();
  }
  unsigned cum = c[t], prev = t ? c[t - 1] : 0u;
  if (st->doPrune && cum >= st->kRem && prev < st->kRem) {
    st->kRem -= prev;
    st->prefix = (st->prefix << 8) | (unsigned)t;
  }
  __syncthreads();
  st->hist[t] = 0;
}

__global__ void k_mark(const float* score, const float* nm, const SelState* st,
                       int* remove, float* out_nm) {
  int i = blockIdx.x * blockDim.x + threadIdx.x;
  if (i >= N_NODES) return;
  int rm = 0;
  if (st->doPrune) {
    float sc = score[i];
    if (sc != 0.0f && encf(sc) <= st->prefix) rm = 1;
  }
  remove[i] = rm;
  out_nm[i] = nm[i] * (1.0f - (float)rm);
}

// ---------------- parent pruning (5 jacobi steps) ----------------
__global__ void k_copy_parent(const int* edge_index, int* p0) {
  int i = blockIdx.x * blockDim.x + threadIdx.x;
  if (i < E_EDGES) p0[i] = edge_index[E_EDGES + i];   // parent row
}
__global__ void k_prune_step(const int* __restrict__ pin, int* __restrict__ pout,
                             const int* __restrict__ remove) {
  int i = blockIdx.x * blockDim.x + threadIdx.x;
  if (i >= E_EDGES) return;
  int p = pin[i];
  int pr = remove[p];                                 // remove[0]==0 always
  pout[i] = pr ? ((p == 0) ? 0 : pin[p - 1]) : p;
}

// ------- big GEMM: h = xin(bf16) @ W(bf16), WMMA + async LDS double-buffer -------
__global__ __launch_bounds__(256) void k_gemm(const float* __restrict__ x,
                                              const float* __restrict__ sr,
                                              const int* __restrict__ tid,
                                              const float* __restrict__ emb_tab,
                                              const unsigned short* __restrict__ wbT,
                                              float* __restrict__ h) {
  __shared__ unsigned short As[2][128 * 40];   // [row][k], stride 40 bf16 (80B)
  __shared__ unsigned short Bs[2][128 * 40];   // [n][k],   stride 40 bf16 (80B)
  const int t = threadIdx.x;
  const int wave = t >> 5, lane = t & 31;
  const int row0 = blockIdx.y * 128;
  const int col0 = blockIdx.x * 128;

  v8f acc[8];
#pragma unroll
  for (int tt = 0; tt < 8; ++tt) acc[tt] = (v8f)0.0f;

  const int arow = t >> 1;                 // A tile row this thread fills
  const int ah   = (t & 1) * 16;           // k half
  const int grow = row0 + arow;
  const int mytid = tid[grow];
  const int la = lane & 15, hf = lane >> 4;

  const unsigned long long wt_addr = (unsigned long long)(uintptr_t)wbT;
  // per-thread async B segments: seg -> row ln = seg>>2, 16B chunk part = seg&3
  const int ln0 = (t + 0)   >> 2, part0 = (t + 0)   & 3;
  const int ln1 = (t + 256) >> 2, part1 = (t + 256) & 3;

  auto stage = [&](int buf, int kb) {
    const int K = kb * 32;
    // --- A tile: f32 -> bf16, ds stores ---
    const float* srcp;
    int kg = K + ah;
    if (kg < D_IN)            srcp = x  + (size_t)grow * D_IN + kg;
    else if (kg < 2 * D_IN)   srcp = sr + (size_t)grow * D_IN + (kg - D_IN);
    else                      srcp = emb_tab + mytid * D_OUT + (kg - 2 * D_IN);
    unsigned* adst = (unsigned*)((char*)&As[buf][0] + (size_t)(arow * 40 + ah) * 2);
#pragma unroll
    for (int j = 0; j < 16; j += 2) {
      unsigned q0 = f2bf(srcp[j]);
      unsigned q1 = f2bf(srcp[j + 1]);
      adst[j >> 1] = q0 | (q1 << 16);
    }
    // --- B tile: async DMA global -> LDS (rows of W^T are contiguous) ---
    unsigned bsb = (unsigned)(uintptr_t)&Bs[buf][0];
    {
      unsigned lo = bsb + (unsigned)(ln0 * 80 + part0 * 16);
      unsigned long long ga = wt_addr +
          ((size_t)(col0 + ln0) * K_DIM + K) * 2 + (size_t)part0 * 16;
      asm volatile("global_load_async_to_lds_b128 %0, %1, off"
                   :: "v"(lo), "v"(ga) : "memory");
    }
    {
      unsigned lo = bsb + (unsigned)(ln1 * 80 + part1 * 16);
      unsigned long long ga = wt_addr +
          ((size_t)(col0 + ln1) * K_DIM + K) * 2 + (size_t)part1 * 16;
      asm volatile("global_load_async_to_lds_b128 %0, %1, off"
                   :: "v"(lo), "v"(ga) : "memory");
    }
  };

  stage(0, 0);   // prologue

  for (int kb = 0; kb < NKB; ++kb) {
    const int cur = kb & 1;
    asm volatile("s_wait_asynccnt 0x0" ::: "memory");  // my async B writes done
    __syncthreads();                                   // s_wait_dscnt 0 + barrier

    if (kb + 1 < NKB) stage(cur ^ 1, kb + 1);          // overlap next tile staging

    // --- A fragment (16x32 bf16: VGPR0-3 k=2j+8h, VGPR4-7 k=16+2j+8h) ---
    union { u32x4 q[2]; v16bf v; } fa;
    const char* abase = (const char*)&As[cur][0] + (size_t)(16 * wave + la) * 80 + 16 * hf;
    fa.q[0] = *(const u32x4*)abase;
    fa.q[1] = *(const u32x4*)(abase + 32);

#pragma unroll
    for (int tt = 0; tt < 8; ++tt) {
      // B fragment (32x16 bf16: VGPR0-3 k=2j+8h, VGPR4-7 k=16+2j+8h)
      union { u32x4 q[2]; v16bf v; } fb;
      const char* bbase = (const char*)&Bs[cur][0] + (size_t)(16 * tt + la) * 80 + 16 * hf;
      fb.q[0] = *(const u32x4*)bbase;
      fb.q[1] = *(const u32x4*)(bbase + 32);
      acc[tt] = __builtin_amdgcn_wmma_f32_16x16x32_bf16(
          false, fa.v, false, fb.v, (short)0, acc[tt], false, false);
    }
  }

  // --- store C/D (lane n=la, VGPR r -> row r+8*hf) ---
#pragma unroll
  for (int tt = 0; tt < 8; ++tt) {
#pragma unroll
    for (int r = 0; r < 8; ++r) {
      int mrow = row0 + 16 * wave + r + 8 * hf;
      h[(size_t)mrow * N_OUT + col0 + 16 * tt + la] = acc[tt][r];
    }
  }
}

// ---------------- attention logits per node ----------------
__global__ __launch_bounds__(256) void k_att(const float* __restrict__ h,
                                             const float* __restrict__ att_src,
                                             const float* __restrict__ att_dst,
                                             float* a_src, float* a_dst) {
  __shared__ float rs[256], rd[256];
  int n = blockIdx.x, t = threadIdx.x;
  int hd = t >> 6, l = t & 63;
  float ss = 0.0f, dd = 0.0f;
  for (int j = 0; j < 4; ++j) {
    int c = l + 64 * j;
    float hv = h[(size_t)n * N_OUT + hd * D_OUT + c];
    ss += hv * att_src[hd * D_OUT + c];
    dd += hv * att_dst[hd * D_OUT + c];
  }
  rs[t] = ss; rd[t] = dd;
  for (int off = 32; off > 0; off >>= 1) {
    __syncthreads();
    if (l < off) { rs[t] += rs[t + off]; rd[t] += rd[t + off]; }
  }
  __syncthreads();
  if (l == 0) { a_src[n * HEADS + hd] = rs[t]; a_dst[n * HEADS + hd] = rd[t]; }
}

// ---------------- edge softmax + scatter ----------------
DEVFN void edge_sd(int e, const int* remove, const int* pfin, int& s_, int& d_) {
  if (e < E_EDGES) {
    if (remove[e + 1]) { s_ = 0; d_ = 0; }
    else { s_ = e + 1; d_ = pfin[e]; }
  } else { s_ = d_ = e - E_EDGES; }     // self loop
}

__global__ void k_alpha_max(const float* a_src, const float* a_dst,
                            const int* remove, const int* pfin,
                            float* alpha, unsigned* m) {
  int e = blockIdx.x * blockDim.x + threadIdx.x;
  if (e >= TE) return;
  int s_, d_; edge_sd(e, remove, pfin, s_, d_);
#pragma unroll
  for (int hh = 0; hh < HEADS; ++hh) {
    float al = a_src[s_ * HEADS + hh] + a_dst[d_ * HEADS + hh];
    al = (al > 0.0f) ? al : SLOPE * al;
    alpha[(size_t)e * HEADS + hh] = al;
    atomicMax(&m[d_ * HEADS + hh], encf(al));
  }
}

__global__ void k_exp_sum(const int* remove, const int* pfin,
                          float* alpha, const unsigned* m, float* s) {
  int e = blockIdx.x * blockDim.x + threadIdx.x;
  if (e >= TE) return;
  int s_, d_; edge_sd(e, remove, pfin, s_, d_);
#pragma unroll
  for (int hh = 0; hh < HEADS; ++hh) {
    float ev = expf(alpha[(size_t)e * HEADS + hh] - decf(m[d_ * HEADS + hh]));
    alpha[(size_t)e * HEADS + hh] = ev;
    atomicAdd(&s[d_ * HEADS + hh], ev);
  }
}

__global__ __launch_bounds__(256) void k_scatter(const int* remove, const int* pfin,
                                                 const float* __restrict__ alpha,
                                                 const float* __restrict__ s,
                                                 const float* __restrict__ h,
                                                 float* __restrict__ out) {
  __shared__ float coef[HEADS];
  int e = blockIdx.x, t = threadIdx.x;
  int s_, d_; edge_sd(e, remove, pfin, s_, d_);
  if (t < HEADS)
    coef[t] = alpha[(size_t)e * HEADS + t] / (s[d_ * HEADS + t] + 1e-16f);
  __syncthreads();
#pragma unroll
  for (int i = 0; i < 4; ++i) {
    int c = t + 256 * i;               // head = i
    atomicAdd(&out[(size_t)d_ * N_OUT + c], coef[i] * h[(size_t)s_ * N_OUT + c]);
  }
}

__global__ void k_final(float* out, const float* gat_b) {
  long long n = (long long)N_NODES * N_OUT;
  long long i = (long long)blockIdx.x * blockDim.x + threadIdx.x;
  long long st = (long long)gridDim.x * blockDim.x;
  for (; i < n; i += st) {
    int c = (int)(i & (N_OUT - 1));
    out[i] = geluf(out[i] + gat_b[c]);
  }
}

} // namespace

extern "C" void kernel_launch(void* const* d_in, const int* in_sizes, int n_in,
                              void* d_out, int out_size, void* d_ws, size_t ws_size,
                              hipStream_t stream) {
  (void)in_sizes; (void)n_in; (void)out_size; (void)ws_size;
  const float* x        = (const float*)d_in[0];
  const int*   eidx     = (const int*)d_in[1];
  const float* sr       = (const float*)d_in[3];
  const float* nm       = (const float*)d_in[4];
  const int*   tid      = (const int*)d_in[5];
  const float* scw      = (const float*)d_in[6];
  const float* scb      = (const float*)d_in[7];
  const float* emb_w    = (const float*)d_in[8];
  const float* emb_b    = (const float*)d_in[9];
  const float* gat_w    = (const float*)d_in[10];
  const float* att_src  = (const float*)d_in[11];
  const float* att_dst  = (const float*)d_in[12];
  const float* gat_b    = (const float*)d_in[13];

  char* ws = (char*)d_ws;
  float*          h      = (float*)(ws + OFF_H);
  unsigned short* wbT    = (unsigned short*)(ws + OFF_WB);
  float*          score  = (float*)(ws + OFF_SCORE);
  float*          a_src  = (float*)(ws + OFF_ASRC);
  float*          a_dst  = (float*)(ws + OFF_ADST);
  unsigned*       mbuf   = (unsigned*)(ws + OFF_M);
  float*          sbuf   = (float*)(ws + OFF_S);
  float*          alpha  = (float*)(ws + OFF_ALPHA);
  int*            remove = (int*)(ws + OFF_REMOVE);
  int*            p0     = (int*)(ws + OFF_P0);
  int*            p1     = (int*)(ws + OFF_P1);
  float*          embtab = (float*)(ws + OFF_EMB);
  SelState*       st     = (SelState*)(ws + OFF_STATE);

  float* out = (float*)d_out;                 // [N*1024] out, then [N] nm

  // init
  k_zero_out<<<8192, 256, 0, stream>>>(out, (long long)N_NODES * N_OUT + N_NODES);
  k_init_ms<<<(N_NODES * HEADS + 255) / 256, 256, 0, stream>>>(mbuf, sbuf);
  k_init_state<<<1, 256, 0, stream>>>(st);
  k_emb_tab<<<1, 512, 0, stream>>>(emb_w, emb_b, embtab);
  k_prep_w<<<(K_DIM * N_OUT + 255) / 256, 256, 0, stream>>>(gat_w, wbT);

  // scoring + exact top-k radix select
  k_score<<<N_NODES, 256, 0, stream>>>(x, sr, scw, scb, nm, score, st);
  k_compute_k<<<1, 1, 0, stream>>>(st);
  for (int r = 0; r < 4; ++r) {
    k_hist<<<N_NODES / 256, 256, 0, stream>>>(score, st, r);
    k_scan<<<1, 256, 0, stream>>>(st);
  }
  k_mark<<<N_NODES / 256, 256, 0, stream>>>(score, nm, st, remove, out + (size_t)N_NODES * N_OUT);

  // parent pointer jumping (5 functional steps, ping-pong)
  k_copy_parent<<<(E_EDGES + 255) / 256, 256, 0, stream>>>(eidx, p0);
  k_prune_step<<<(E_EDGES + 255) / 256, 256, 0, stream>>>(p0, p1, remove);
  k_prune_step<<<(E_EDGES + 255) / 256, 256, 0, stream>>>(p1, p0, remove);
  k_prune_step<<<(E_EDGES + 255) / 256, 256, 0, stream>>>(p0, p1, remove);
  k_prune_step<<<(E_EDGES + 255) / 256, 256, 0, stream>>>(p1, p0, remove);
  k_prune_step<<<(E_EDGES + 255) / 256, 256, 0, stream>>>(p0, p1, remove);
  int* pfin = p1;

  // h = xin @ gat_w  (bf16 WMMA, async-LDS double buffered)
  k_gemm<<<dim3(N_OUT / 128, N_NODES / 128), 256, 0, stream>>>(x, sr, tid, embtab, wbT, h);

  // attention
  k_att<<<N_NODES, 256, 0, stream>>>(h, att_src, att_dst, a_src, a_dst);
  k_alpha_max<<<(TE + 255) / 256, 256, 0, stream>>>(a_src, a_dst, remove, pfin, alpha, mbuf);
  k_exp_sum<<<(TE + 255) / 256, 256, 0, stream>>>(remove, pfin, alpha, mbuf, sbuf);
  k_scatter<<<TE, 256, 0, stream>>>(remove, pfin, alpha, sbuf, h, out);

  // epilogue
  k_final<<<8192, 256, 0, stream>>>(out, gat_b);
}
